// MultiHeadAttentionAwareTemporalContex_qc_k1d_17489106829907
// MI455X (gfx1250) — matrix operand order repaired and checked by reading, not compiled, not run on hardware
//
#include <hip/hip_runtime.h>
#include <hip/hip_bf16.h>

// MI455X (gfx1250) fused conv-attention:
//   qc = causal conv3(query), kc = same-pad conv3(key), v = value @ WvT + b
//   per-head causal softmax attention (T=64, DK=16), out = ctx @ WoT + b
// One workgroup (8 wave32) per (b,n) slab; all GEMMs via v_wmma_f32_16x16x32_bf16.

typedef __bf16 bf16_t;
typedef __attribute__((ext_vector_type(16))) __bf16 v16bf;
typedef __attribute__((ext_vector_type(8)))  __bf16 v8bf;
typedef __attribute__((ext_vector_type(8)))  float  v8f;

#define B_  8
#define N_  512
#define T_  64
#define D_  128
#define H_  8
#define DK_ 16

// ---------- fragment helpers (layouts per CDNA5 ISA 7.12.2) ----------

static __device__ __forceinline__ v8bf zero8() {
  v8bf z;
#pragma unroll
  for (int i = 0; i < 8; ++i) z[i] = (bf16_t)0.0f;
  return z;
}

static __device__ __forceinline__ v16bf combine16(v8bf lo, v8bf hi) {
  v16bf r;
#pragma unroll
  for (int i = 0; i < 8; ++i) { r[i] = lo[i]; r[i + 8] = hi[i]; }
  return r;
}

// A fragment (16xK=32 bf16) from row-major src[rows x ld].
// Lane layout: row = row0 + lane%16 + shift; half i -> K = (i<8 ? i+8g : i+8+8g).
// Rows outside [0,nrows) read as zero (conv boundary).
static __device__ __forceinline__ v16bf load_a32(const bf16_t* src, int ld,
                                                 int row0, int k0, int shift,
                                                 int nrows) {
  const int lane = threadIdx.x & 31;
  const int g = lane >> 4;
  const int row = row0 + (lane & 15) + shift;
  v8bf lo = zero8(), hi = zero8();
  if (row >= 0 && row < nrows) {
    const bf16_t* p = src + row * ld + k0;
    lo = *(const v8bf*)(p + 8 * g);        // K = k0 + 8g .. +7
    hi = *(const v8bf*)(p + 16 + 8 * g);   // K = k0 + 16 + 8g .. +7
  }
  return combine16(lo, hi);
}

// A fragment with only 16 real K values (scores GEMM, DK=16 padded to K=32).
static __device__ __forceinline__ v16bf load_a16(const bf16_t* src, int ld,
                                                 int row0, int col0) {
  const int lane = threadIdx.x & 31;
  const int g = lane >> 4;
  const int row = row0 + (lane & 15);
  v8bf lo = *(const v8bf*)(src + row * ld + col0 + 8 * g);  // K = 8g..8g+7
  return combine16(lo, zero8());                            // K>=16 -> 0
}

// B fragment (K=32 x N=16) where B[kk][n] = w[(col0+n)*ld + k0 + kk].
// Lane layout: n = col0 + lane%16; half i -> K = k0 + (i<8 ? i+8g : i+8+8g).
static __device__ __forceinline__ v16bf load_bT(const bf16_t* w, int ld,
                                                int col0, int k0) {
  const int lane = threadIdx.x & 31;
  const int g = lane >> 4;
  const bf16_t* p = w + (col0 + (lane & 15)) * ld + k0 + 8 * g;
  return combine16(*(const v8bf*)p, *(const v8bf*)(p + 16));
}

// B fragment with 16 real K values: B[kk][n] = src[(row_n0+n)*ld + kcol0 + kk], kk<16.
static __device__ __forceinline__ v16bf load_bT16(const bf16_t* src, int ld,
                                                  int row_n0, int kcol0) {
  const int lane = threadIdx.x & 31;
  const int g = lane >> 4;
  v8bf lo = *(const v8bf*)(src + (row_n0 + (lane & 15)) * ld + kcol0 + 8 * g);
  return combine16(lo, zero8());
}

static __device__ __forceinline__ v8f wmma_bf16(v16bf a, v16bf b, v8f c) {
  return __builtin_amdgcn_wmma_f32_16x16x32_bf16(false, a, false, b,
                                                 (short)0, c, false, false);
}

// C/D layout: VGPR r holds (row0 + r + 8g, col0 + lane%16).
static __device__ __forceinline__ void store_c_bf16(bf16_t* dst, int ld,
                                                    int row0, int col0, v8f c) {
  const int lane = threadIdx.x & 31;
  const int g = lane >> 4, n = lane & 15;
#pragma unroll
  for (int r = 0; r < 8; ++r)
    dst[(row0 + r + 8 * g) * ld + col0 + n] = (bf16_t)c[r];
}

static __device__ __forceinline__ void store_c_bf16_T(bf16_t* dst, int ld,
                                                      int row0, int col0, v8f c) {
  const int lane = threadIdx.x & 31;
  const int g = lane >> 4, n = lane & 15;
#pragma unroll
  for (int r = 0; r < 8; ++r)
    dst[(col0 + n) * ld + row0 + r + 8 * g] = (bf16_t)c[r];
}

// ---------- weight prep: f32 -> bf16, laid out [n][k] for contiguous B-frags ----
// ws layout (bf16 elems): wq[3][128][128] @0, wk @49152, wv[128][128] @98304, wo @114688
__global__ void prep_weights(const float* __restrict__ qw, const float* __restrict__ kw,
                             const float* __restrict__ vw, const float* __restrict__ ow,
                             bf16_t* __restrict__ ws) {
  int idx = blockIdx.x * blockDim.x + threadIdx.x;
  if (idx < 49152) {                       // q_conv_w (dout,din,1,K) -> [k][dout][din]
    int k = idx / 16384, r = idx % 16384;
    ws[idx] = (bf16_t)qw[r * 3 + k];       // r = dout*128 + din
  } else if (idx < 98304) {
    int j = idx - 49152;
    int k = j / 16384, r = j % 16384;
    ws[idx] = (bf16_t)kw[r * 3 + k];
  } else if (idx < 114688) {
    ws[idx] = (bf16_t)vw[idx - 98304];     // already (dout,din)
  } else if (idx < 131072) {
    ws[idx] = (bf16_t)ow[idx - 114688];
  }
}

// ---------- fused attention kernel: one block per (b,n) slab ----------
__global__ __launch_bounds__(256) void attn_fused(
    const float* __restrict__ q_in, const float* __restrict__ k_in,
    const float* __restrict__ v_in,
    const float* __restrict__ qb, const float* __restrict__ kbias,
    const float* __restrict__ vb, const float* __restrict__ ob,
    const bf16_t* __restrict__ wts, float* __restrict__ out) {
  // 80 KB dynamic LDS (hardware: 320 KB/WGP):
  //   sX[64*128] staging(q/k/v in) then ctx | sQ | sK | sVt[128][64] | sP[8][16*64]
  extern __shared__ __align__(16) bf16_t smem[];
  bf16_t* sX  = smem;
  bf16_t* sQ  = smem + 8192;
  bf16_t* sK  = smem + 16384;
  bf16_t* sVt = smem + 24576;
  bf16_t* sP  = smem + 32768;

  const bf16_t* wq = wts;
  const bf16_t* wk = wts + 49152;
  const bf16_t* wv = wts + 98304;
  const bf16_t* wo = wts + 114688;

  const int tid  = threadIdx.x;
  const int wave = tid >> 5;
  const int lane = tid & 31;
  const int g    = lane >> 4;
  const int nn   = lane & 15;
  const size_t slab = (size_t)blockIdx.x * (T_ * D_);

  const int t0w = 16 * (wave >> 1);  // projection strip: rows
  const int c0w = 64 * (wave & 1);   // projection strip: column half

  // ---- stage 1: qc = causal conv(query) ----
  for (int i = tid; i < T_ * D_; i += 256) sX[i] = (bf16_t)q_in[slab + i];
  __syncthreads();
  {
    v8f acc[4];
#pragma unroll
    for (int jt = 0; jt < 4; ++jt) {
      float bv = qb[c0w + 16 * jt + nn];
#pragma unroll
      for (int r = 0; r < 8; ++r) acc[jt][r] = bv;
    }
#pragma unroll
    for (int tap = 0; tap < 3; ++tap) {
      const bf16_t* wtap = wq + tap * (D_ * D_);
#pragma unroll
      for (int kb = 0; kb < 4; ++kb) {
        v16bf a = load_a32(sX, D_, t0w, 32 * kb, tap - 2, T_);
#pragma unroll
        for (int jt = 0; jt < 4; ++jt)
          acc[jt] = wmma_bf16(a, load_bT(wtap, D_, c0w + 16 * jt, 32 * kb), acc[jt]);
      }
    }
#pragma unroll
    for (int jt = 0; jt < 4; ++jt) store_c_bf16(sQ, D_, t0w, c0w + 16 * jt, acc[jt]);
  }
  __syncthreads();

  // ---- stage 2: kc = same-pad conv(key) ----
  for (int i = tid; i < T_ * D_; i += 256) sX[i] = (bf16_t)k_in[slab + i];
  __syncthreads();
  {
    v8f acc[4];
#pragma unroll
    for (int jt = 0; jt < 4; ++jt) {
      float bv = kbias[c0w + 16 * jt + nn];
#pragma unroll
      for (int r = 0; r < 8; ++r) acc[jt][r] = bv;
    }
#pragma unroll
    for (int tap = 0; tap < 3; ++tap) {
      const bf16_t* wtap = wk + tap * (D_ * D_);
#pragma unroll
      for (int kb = 0; kb < 4; ++kb) {
        v16bf a = load_a32(sX, D_, t0w, 32 * kb, tap - 1, T_);
#pragma unroll
        for (int jt = 0; jt < 4; ++jt)
          acc[jt] = wmma_bf16(a, load_bT(wtap, D_, c0w + 16 * jt, 32 * kb), acc[jt]);
      }
    }
#pragma unroll
    for (int jt = 0; jt < 4; ++jt) store_c_bf16(sK, D_, t0w, c0w + 16 * jt, acc[jt]);
  }
  __syncthreads();

  // ---- stage 3: v = value @ WvT + b (stored transposed [d][t]) ----
  for (int i = tid; i < T_ * D_; i += 256) sX[i] = (bf16_t)v_in[slab + i];
  __syncthreads();
  {
    v8f acc[4];
#pragma unroll
    for (int jt = 0; jt < 4; ++jt) {
      float bv = vb[c0w + 16 * jt + nn];
#pragma unroll
      for (int r = 0; r < 8; ++r) acc[jt][r] = bv;
    }
#pragma unroll
    for (int kb = 0; kb < 4; ++kb) {
      v16bf a = load_a32(sX, D_, t0w, 32 * kb, 0, T_);
#pragma unroll
      for (int jt = 0; jt < 4; ++jt)
        acc[jt] = wmma_bf16(a, load_bT(wv, D_, c0w + 16 * jt, 32 * kb), acc[jt]);
    }
#pragma unroll
    for (int jt = 0; jt < 4; ++jt) store_c_bf16_T(sVt, T_, t0w, c0w + 16 * jt, acc[jt]);
  }
  __syncthreads();

  // ---- stage 4: attention, wave = head; ctx written into sX ----
  {
    const int h = wave;
    bf16_t* pw = sP + h * (16 * 64);
#pragma unroll
    for (int is = 0; is < 4; ++is) {
      const int t0 = 16 * is;
      v8f s4[4];
#pragma unroll
      for (int j = 0; j < 4; ++j)
#pragma unroll
        for (int r = 0; r < 8; ++r) s4[j][r] = 0.0f;

      v16bf aq = load_a16(sQ, D_, t0, h * DK_);
#pragma unroll
      for (int j = 0; j < 4; ++j)
        s4[j] = wmma_bf16(aq, load_bT16(sK, D_, 16 * j, h * DK_), s4[j]);

      // scale 1/sqrt(16), causal mask, row softmax (rows live in a 16-lane half)
#pragma unroll
      for (int r = 0; r < 8; ++r) {
        const int trow = t0 + r + 8 * g;
        float mx = -3.0e38f;
#pragma unroll
        for (int j = 0; j < 4; ++j) {
          float v = s4[j][r] * 0.25f;
          if (16 * j + nn > trow) v = -1.0e9f;
          s4[j][r] = v;
          mx = fmaxf(mx, v);
        }
#pragma unroll
        for (int d = 1; d < 16; d <<= 1) mx = fmaxf(mx, __shfl_xor(mx, d, 32));
        float sum = 0.0f;
#pragma unroll
        for (int j = 0; j < 4; ++j) {
          float e = __expf(s4[j][r] - mx);
          s4[j][r] = e;
          sum += e;
        }
#pragma unroll
        for (int d = 1; d < 16; d <<= 1) sum += __shfl_xor(sum, d, 32);
        const float inv = 1.0f / sum;
#pragma unroll
        for (int j = 0; j < 4; ++j)
          pw[(r + 8 * g) * 64 + 16 * j + nn] = (bf16_t)(s4[j][r] * inv);
      }

      // ctx strip (16 x 16) = p(16x64) @ v_h(64x16)
      v8f c;
#pragma unroll
      for (int r = 0; r < 8; ++r) c[r] = 0.0f;
#pragma unroll
      for (int kb = 0; kb < 2; ++kb)
        c = wmma_bf16(load_a32(pw, 64, 0, 32 * kb, 0, 16),
                      load_bT(sVt, T_, h * DK_, 32 * kb), c);
      store_c_bf16(sX, D_, t0, h * DK_, c);  // ctx layout (t, h*DK+dk)
    }
  }
  __syncthreads();

  // ---- stage 5: out = ctx @ WoT + b ----
  {
    v8f acc[4];
#pragma unroll
    for (int jt = 0; jt < 4; ++jt) {
      float bv = ob[c0w + 16 * jt + nn];
#pragma unroll
      for (int r = 0; r < 8; ++r) acc[jt][r] = bv;
    }
#pragma unroll
    for (int kb = 0; kb < 4; ++kb) {
      v16bf a = load_a32(sX, D_, t0w, 32 * kb, 0, T_);
#pragma unroll
      for (int jt = 0; jt < 4; ++jt)
        acc[jt] = wmma_bf16(a, load_bT(wo, D_, c0w + 16 * jt, 32 * kb), acc[jt]);
    }
#pragma unroll
    for (int jt = 0; jt < 4; ++jt)
#pragma unroll
      for (int r = 0; r < 8; ++r)
        out[slab + (size_t)(t0w + r + 8 * g) * D_ + c0w + 16 * jt + nn] = acc[jt][r];
  }
}

extern "C" void kernel_launch(void* const* d_in, const int* in_sizes, int n_in,
                              void* d_out, int out_size, void* d_ws, size_t ws_size,
                              hipStream_t stream) {
  (void)in_sizes; (void)n_in; (void)out_size; (void)ws_size;
  const float* q   = (const float*)d_in[0];
  const float* k   = (const float*)d_in[1];
  const float* v   = (const float*)d_in[2];
  // d_in[3] = causal mask (known tril, unused)
  const float* qw  = (const float*)d_in[4];
  const float* qbv = (const float*)d_in[5];
  const float* kw  = (const float*)d_in[6];
  const float* kbv = (const float*)d_in[7];
  const float* vw  = (const float*)d_in[8];
  const float* vbv = (const float*)d_in[9];
  const float* ow  = (const float*)d_in[10];
  const float* obv = (const float*)d_in[11];
  bf16_t* wts = (bf16_t*)d_ws;  // 131072 bf16 = 256 KB of scratch

  prep_weights<<<512, 256, 0, stream>>>(qw, kw, vw, ow, wts);
  attn_fused<<<B_ * N_, 256, 81920, stream>>>(q, k, v, qbv, kbv, vbv, obv, wts,
                                              (float*)d_out);
}
